// GCN__69114613729586
// MI455X (gfx1250) — matrix operand order repaired and checked by reading, not compiled, not run on hardware
//
#include <hip/hip_runtime.h>
#include <hip/hip_bf16.h>

#define NN 50000
#define FDIM 64
#define NCLS 16
#define TPB 256

typedef __attribute__((ext_vector_type(16))) __bf16 v16bf;
typedef __attribute__((ext_vector_type(16))) float  v16f;
typedef __attribute__((ext_vector_type(8)))  float  v8f;

__device__ __forceinline__ __bf16 f2bf(float f) {
  union { float f; unsigned u; } in; in.f = f;
  unsigned r = (in.u + 0x7FFFu + ((in.u >> 16) & 1u)) >> 16;   // RNE
  union { unsigned short s; __bf16 b; } out; out.s = (unsigned short)r;
  return out.b;
}

__device__ __forceinline__ void atomAddF(float* p, float v) {
  unsafeAtomicAdd(p, v);          // hw global_atomic_add_f32
}

// ---------------- normalization ----------------
__global__ void k_deg_init(float* __restrict__ deg, int n) {
  int i = blockIdx.x * blockDim.x + threadIdx.x;
  if (i < n) deg[i] = 1.0f;                      // self-loop weight
}

__global__ void k_deg_scatter(const int* __restrict__ col, const float* __restrict__ w,
                              float* __restrict__ deg, int e) {
  int i = blockIdx.x * blockDim.x + threadIdx.x;
  if (i < e) atomAddF(&deg[col[i]], w[i]);
}

__global__ void k_dinv(float* __restrict__ deg, int n) {
  int i = blockIdx.x * blockDim.x + threadIdx.x;
  if (i < n) { float d = deg[i]; deg[i] = (d > 0.0f) ? rsqrtf(d) : 0.0f; }
}

__global__ void k_norm(const int* __restrict__ row, const int* __restrict__ col,
                       const float* __restrict__ w, const float* __restrict__ dinv,
                       float* __restrict__ norm, int e) {
  int i = blockIdx.x * blockDim.x + threadIdx.x;
  if (i < e) norm[i] = dinv[row[i]] * w[i] * dinv[col[i]];
}

// ---------------- weight packing into WMMA B layout ----------------
// B (32x16 bf16 per step): lane holds column n = lane&15; lanes 0-15 carry
// K=kb*32+{0..7,16..23}, lanes 16-31 carry K=kb*32+{8..15,24..31}.
__global__ void k_pack_w(const float* __restrict__ W, __bf16* __restrict__ Wp,
                         int nbCount, int outw) {
  int idx = blockIdx.x * blockDim.x + threadIdx.x;
  int total = nbCount * 2 * 32 * 16;
  if (idx >= total) return;
  int i    = idx & 15;
  int lane = (idx >> 4) & 31;
  int blk  = idx >> 9;                // nb*2 + kb
  int kb = blk & 1, nb = blk >> 1;
  int k = kb * 32 + ((lane < 16) ? 0 : 8) + ((i < 8) ? i : i + 8);
  int c = nb * 16 + (lane & 15);
  Wp[idx] = f2bf(W[k * outw + c]);
}

// ---------------- dense projection via bf16 WMMA ----------------
// One wave -> 16 rows x (NB*16) cols. A is fp32 [nrows x 64] converted on load
// (optionally through ReLU); K=64 split into two 16x16x32 steps.
template <int NB, bool RELU>
__global__ void k_gemm(const float* __restrict__ src, const __bf16* __restrict__ Wp,
                       float* __restrict__ dst, int nrows) {
  const int lane = threadIdx.x & 31;
  const int wid  = (blockIdx.x * blockDim.x + threadIdx.x) >> 5;
  const int row0 = wid * 16;
  if (row0 >= nrows) return;                     // uniform per wave

  int rowa = row0 + (lane & 15);
  if (rowa > nrows - 1) rowa = nrows - 1;
  const int  off0 = (lane < 16) ? 0 : 8;         // 16-bit A layout (interleaved K)
  const float* p = src + (size_t)rowa * FDIM;

  v16bf A[2];
#pragma unroll
  for (int kb = 0; kb < 2; ++kb) {
    const int base = kb * 32 + off0;
    float4 q0 = *(const float4*)(p + base + 0);
    float4 q1 = *(const float4*)(p + base + 4);
    float4 q2 = *(const float4*)(p + base + 16);
    float4 q3 = *(const float4*)(p + base + 20);
    v16f t;
    t[0]=q0.x;  t[1]=q0.y;  t[2]=q0.z;  t[3]=q0.w;
    t[4]=q1.x;  t[5]=q1.y;  t[6]=q1.z;  t[7]=q1.w;
    t[8]=q2.x;  t[9]=q2.y;  t[10]=q2.z; t[11]=q2.w;
    t[12]=q3.x; t[13]=q3.y; t[14]=q3.z; t[15]=q3.w;
    if (RELU) {
#pragma unroll
      for (int i = 0; i < 16; ++i) t[i] = fmaxf(t[i], 0.0f);
    }
    A[kb] = __builtin_convertvector(t, v16bf);   // vector fptrunc (RNE)
  }

  v8f acc[NB];
#pragma unroll
  for (int nb = 0; nb < NB; ++nb) { v8f z = {}; acc[nb] = z; }

#pragma unroll
  for (int nb = 0; nb < NB; ++nb) {
#pragma unroll
    for (int kb = 0; kb < 2; ++kb) {
      v16bf B = *(const v16bf*)(Wp + ((size_t)(nb * 2 + kb) * 32 + lane) * 16);
      acc[nb] = __builtin_amdgcn_wmma_f32_16x16x32_bf16(
          false, A[kb], false, B, (short)0, acc[nb], false, false);
    }
  }

  const int rbase = row0 + ((lane < 16) ? 0 : 8);  // C/D layout
  const int c0    = lane & 15;
  if (row0 + 16 <= nrows) {                        // uniform fast path, no guards
#pragma unroll
    for (int nb = 0; nb < NB; ++nb)
#pragma unroll
      for (int r = 0; r < 8; ++r)
        dst[(size_t)(rbase + r) * (NB * 16) + nb * 16 + c0] = acc[nb][r];
  } else {                                         // generic tail (unused for N%16==0)
#pragma unroll
    for (int nb = 0; nb < NB; ++nb)
#pragma unroll
      for (int r = 0; r < 8; ++r) {
        int rr = rbase + r;
        if (rr < nrows) dst[(size_t)rr * (NB * 16) + nb * 16 + c0] = acc[nb][r];
      }
  }
}

// ---------------- message passing ----------------
// agg = dinv^2 * h + bias   (self-loop term + bias folded together)
__global__ void k_agg_init(const float* __restrict__ h, const float* __restrict__ dinv,
                           const float* __restrict__ bias, float* __restrict__ agg, int n) {
  int idx = blockIdx.x * blockDim.x + threadIdx.x;
  if (idx >= n * FDIM) return;
  int i = idx >> 6, f = idx & 63;
  float d = dinv[i];
  agg[idx] = d * d * h[idx] + bias[f];
}

// 64 threads per edge: coalesced gather of h[row], fp32 atomic scatter to agg[col]
__global__ void k_scatter(const float* __restrict__ h, const int* __restrict__ row,
                          const int* __restrict__ col, const float* __restrict__ norm,
                          float* __restrict__ agg, int e) {
  int idx = blockIdx.x * blockDim.x + threadIdx.x;
  if (idx >= e * 64) return;                      // 51.2M, fits int32
  int eid = idx >> 6, f = idx & 63;
  float m = h[(size_t)row[eid] * FDIM + f] * norm[eid];
  atomAddF(&agg[(size_t)col[eid] * FDIM + f], m);
}

// ---------------- softmax ----------------
__global__ void k_softmax(const float* __restrict__ logits, const float* __restrict__ bout,
                          float* __restrict__ out, int n) {
  int i = blockIdx.x * blockDim.x + threadIdx.x;
  if (i >= n) return;
  float v[NCLS];
  float mx = -3.4e38f;
#pragma unroll
  for (int c = 0; c < NCLS; ++c) {
    v[c] = logits[i * NCLS + c] + bout[c];
    mx = fmaxf(mx, v[c]);
  }
  float s = 0.0f;
#pragma unroll
  for (int c = 0; c < NCLS; ++c) { v[c] = __expf(v[c] - mx); s += v[c]; }
  float inv = 1.0f / s;
#pragma unroll
  for (int c = 0; c < NCLS; ++c) out[i * NCLS + c] = v[c] * inv;
}

// ---------------- driver ----------------
extern "C" void kernel_launch(void* const* d_in, const int* in_sizes, int n_in,
                              void* d_out, int out_size, void* d_ws, size_t ws_size,
                              hipStream_t stream) {
  const float* x    = (const float*)d_in[0];
  const int*   ei   = (const int*)  d_in[1];
  const float* ew   = (const float*)d_in[2];
  const float* W1   = (const float*)d_in[3];
  const float* b1   = (const float*)d_in[4];
  const float* W2   = (const float*)d_in[5];
  const float* b2   = (const float*)d_in[6];
  const float* Wout = (const float*)d_in[7];
  const float* bout = (const float*)d_in[8];
  float* out = (float*)d_out;

  const int N = NN;
  const int E = in_sizes[2];          // edge_weight length
  const int* row = ei;                // edge_index[0]
  const int* col = ei + E;            // edge_index[1]

  // carve workspace (256B aligned slices)
  char* wsp = (char*)d_ws;
  auto carve = [&](size_t bytes) -> char* {
    char* p = wsp; wsp += (bytes + 255) & ~(size_t)255; return p;
  };
  float*  dinv   = (float*) carve((size_t)N * 4);
  float*  norm   = (float*) carve((size_t)E * 4);
  float*  h      = (float*) carve((size_t)N * FDIM * 4);
  float*  agg    = (float*) carve((size_t)N * FDIM * 4);
  float*  logits = (float*) carve((size_t)N * NCLS * 4);
  __bf16* W1p    = (__bf16*)carve(4 * 2 * 32 * 16 * 2);
  __bf16* W2p    = (__bf16*)carve(4 * 2 * 32 * 16 * 2);
  __bf16* Wop    = (__bf16*)carve(1 * 2 * 32 * 16 * 2);

  auto cdiv = [](long long a, long long b) { return (int)((a + b - 1) / b); };

  // gcn_norm
  k_deg_init   <<<cdiv(N, TPB), TPB, 0, stream>>>(dinv, N);
  k_deg_scatter<<<cdiv(E, TPB), TPB, 0, stream>>>(col, ew, dinv, E);
  k_dinv       <<<cdiv(N, TPB), TPB, 0, stream>>>(dinv, N);
  k_norm       <<<cdiv(E, TPB), TPB, 0, stream>>>(row, col, ew, dinv, norm, E);

  // pack weights to bf16 WMMA layout
  k_pack_w<<<cdiv(4 * 1024, TPB), TPB, 0, stream>>>(W1,   W1p, 4, FDIM);
  k_pack_w<<<cdiv(4 * 1024, TPB), TPB, 0, stream>>>(W2,   W2p, 4, FDIM);
  k_pack_w<<<cdiv(1 * 1024, TPB), TPB, 0, stream>>>(Wout, Wop, 1, NCLS);

  const int gemmGrid = cdiv((long long)(N / 16) * 32, TPB);   // 3125 waves

  // layer 1
  k_gemm<4, false><<<gemmGrid, TPB, 0, stream>>>(x, W1p, h, N);
  k_agg_init<<<cdiv((long long)N * FDIM, TPB), TPB, 0, stream>>>(h, dinv, b1, agg, N);
  k_scatter <<<cdiv((long long)E * 64,  TPB), TPB, 0, stream>>>(h, row, col, norm, agg, E);

  // layer 2 (ReLU fused into A-load of the GEMM)
  k_gemm<4, true><<<gemmGrid, TPB, 0, stream>>>(agg, W2p, h, N);
  k_agg_init<<<cdiv((long long)N * FDIM, TPB), TPB, 0, stream>>>(h, dinv, b2, agg, N);
  k_scatter <<<cdiv((long long)E * 64,  TPB), TPB, 0, stream>>>(h, row, col, norm, agg, E);

  // output projection (ReLU fused) + softmax
  k_gemm<1, true><<<gemmGrid, TPB, 0, stream>>>(agg, Wop, logits, N);
  k_softmax<<<cdiv(N, TPB), TPB, 0, stream>>>(logits, bout, out, N);
}
